// MultiheadAttention_10917806866948
// MI455X (gfx1250) — matrix operand order repaired and checked by reading, not compiled, not run on hardware
//
#include <hip/hip_runtime.h>
#include <hip/hip_bf16.h>

// MHA with LSA (masked diagonal) for MI455X / gfx1250, wave32, f16 WMMA w/ f32 accum.
// B=4, S=2048, D=1024, H=16, DH=64.
// Workspace (f16): Q[B,H,S,DH] | K[B,H,S,DH] | V^T[B,H,DH,S] | A[B,S,D] = 64 MB.

#define BB 4
#define SS 2048
#define DD 1024
#define HH 16
#define DHH 64
#define LSA_NEG -987654321.0f

typedef __attribute__((ext_vector_type(16))) _Float16 v16h;
typedef __attribute__((ext_vector_type(8)))  _Float16 v8h;
typedef __attribute__((ext_vector_type(8)))  float    v8f;
typedef int i32x4 __attribute__((vector_size(16)));

static __device__ __forceinline__ v16h join8(v8h lo, v8h hi) {
  return __builtin_shufflevector(lo, hi, 0,1,2,3,4,5,6,7,8,9,10,11,12,13,14,15);
}
static __device__ __forceinline__ v8f wmma16(v16h a, v16h b, v8f c) {
  return __builtin_amdgcn_wmma_f32_16x16x32_f16(false, a, false, b, (short)0, c, false, false);
}

// ---- gfx1250 async global->LDS copy (ASYNCcnt), with portable fallback ------
#if __has_builtin(__builtin_amdgcn_global_load_async_to_lds_b128)
#define ASYNC_LDS 1
#endif

static __device__ __forceinline__ void cp16_g2l(const _Float16* g, _Float16* l) {
#ifdef ASYNC_LDS
  __builtin_amdgcn_global_load_async_to_lds_b128(
      (__attribute__((address_space(1))) i32x4*)(_Float16*)g,
      (__attribute__((address_space(3))) i32x4*)l, 0, 0);
#else
  *(v8h*)l = *(const v8h*)g;
#endif
}
static __device__ __forceinline__ void async_wait() {
#ifdef ASYNC_LDS
#if __has_builtin(__builtin_amdgcn_s_wait_asynccnt)
  __builtin_amdgcn_s_wait_asynccnt(0);
#else
  asm volatile("s_wait_asynccnt 0x0" ::: "memory");
#endif
#endif
}

// ---------------------------------------------------------------------------
// Kernel 1: Y = X @ W^T + b, X fp32 [M=B*S, D], W fp32 [D, D], out f16.
// 128x128 tile / block, 256 threads = 8 waves (4x2), each wave 32x64 strip.
// vmode 0: head-split [B,H,S,DH].  vmode 1: transposed [B,H,DH,S] (for V).
// ---------------------------------------------------------------------------
__global__ __launch_bounds__(256)
void qkv_proj_kernel(const float* __restrict__ X, const float* __restrict__ W,
                     const float* __restrict__ bias, _Float16* __restrict__ Out,
                     int vmode)
{
  __shared__ _Float16 As[128][40];  // [m][k], +8 pad, rows 80B (16B aligned)
  __shared__ _Float16 Bs[128][40];  // [n][k] == W row-major slice

  const int tid  = threadIdx.x;
  const int lane = tid & 31;
  const int w    = tid >> 5;        // wave 0..7
  const int wr   = w >> 1;          // M strip (32 rows)
  const int wc   = w & 1;           // N strip (64 cols)
  const int half = lane >> 4;
  const int ml   = lane & 15;
  const int mBase = blockIdx.x * 128;
  const int nBase = blockIdx.y * 128;

  v8f acc[2][4] = {};
  const int rS = tid >> 1;          // staging row 0..127
  const int kC = (tid & 1) * 16;    // staging k-chunk 0/16

  for (int k0 = 0; k0 < DD; k0 += 32) {
    const float* xs = X + (size_t)(mBase + rS) * DD + k0 + kC;
    const float* ws = W + (size_t)(nBase + rS) * DD + k0 + kC;
#pragma unroll
    for (int j = 0; j < 16; ++j) As[rS][kC + j] = (_Float16)xs[j];
#pragma unroll
    for (int j = 0; j < 16; ++j) Bs[rS][kC + j] = (_Float16)ws[j];
    if (k0 + 32 < DD) {             // global_prefetch_b8 for next K tile
      __builtin_prefetch(xs + 32, 0, 1);
      __builtin_prefetch(ws + 32, 0, 1);
    }
    __syncthreads();

    v16h a[2];
#pragma unroll
    for (int mT = 0; mT < 2; ++mT) {
      v8h alo = *(const v8h*)&As[wr*32 + mT*16 + ml][half*8];
      v8h ahi = *(const v8h*)&As[wr*32 + mT*16 + ml][16 + half*8];
      a[mT] = join8(alo, ahi);
    }
#pragma unroll
    for (int nT = 0; nT < 4; ++nT) {
      v8h blo = *(const v8h*)&Bs[wc*64 + nT*16 + ml][half*8];
      v8h bhi = *(const v8h*)&Bs[wc*64 + nT*16 + ml][16 + half*8];
      v16h bf = join8(blo, bhi);
#pragma unroll
      for (int mT = 0; mT < 2; ++mT)
        acc[mT][nT] = wmma16(a[mT], bf, acc[mT][nT]);
    }
    __syncthreads();
  }

#pragma unroll
  for (int nT = 0; nT < 4; ++nT) {
    const int col = nBase + wc*64 + nT*16 + ml;
    const float bv = bias[col];
    const int hh = col >> 6, dd = col & 63;
#pragma unroll
    for (int mT = 0; mT < 2; ++mT) {
#pragma unroll
      for (int i = 0; i < 8; ++i) {
        const int row = mBase + wr*32 + mT*16 + half*8 + i;
        const int b = row >> 11, s = row & (SS - 1);
        const _Float16 v = (_Float16)(acc[mT][nT][i] + bv);
        if (vmode)
          Out[(((size_t)b*HH + hh)*DHH + dd)*SS + s] = v;   // [B,H,DH,S]
        else
          Out[(((size_t)b*HH + hh)*SS + s)*DHH + dd] = v;   // [B,H,S,DH]
      }
    }
  }
}

// ---------------------------------------------------------------------------
// Kernel 2: flash-style LSA attention. Block = (b, h, 128-query tile),
// 256 threads = 8 waves, each wave owns 16 query rows. K/V tiles of 64 keys.
// V comes pre-transposed [B,H,DH,S] so all staging is contiguous tile copies
// (async global->LDS when available).
// ---------------------------------------------------------------------------
__global__ __launch_bounds__(256)
void lsa_attn_kernel(const _Float16* __restrict__ Qh, const _Float16* __restrict__ Kh,
                     const _Float16* __restrict__ VTh, _Float16* __restrict__ Aout)
{
  __shared__ _Float16 Qs[128][72];  // [m][d]
  __shared__ _Float16 Ks[64][72];   // [n][d]
  __shared__ _Float16 Vt[64][72];   // [d][n]
  __shared__ _Float16 Ps[128][72];  // [m][n]  wave-private rows

  const int tid  = threadIdx.x;
  const int lane = tid & 31;
  const int w    = tid >> 5;        // 0..7
  const int half = lane >> 4;
  const int ml   = lane & 15;

  const int qb = blockIdx.x & (SS/128 - 1);
  const int bh = blockIdx.x >> 4;          // / (S/128)
  const int h  = bh & (HH - 1);
  const int b  = bh >> 4;
  const size_t base  = ((size_t)(b*HH + h)) * SS * DHH;  // Q/K [.,S,DH]
  const size_t vbase = ((size_t)(b*HH + h)) * DHH * SS;  // V^T [.,DH,S]
  const int sBase = qb * 128;

  { // stage Q tile: 128x64 f16 = 1024 16B chunks, 4 per thread
#pragma unroll
    for (int j = 0; j < 4; ++j) {
      const int cc = tid + 256*j;
      const int r = cc >> 3, co = (cc & 7) * 8;
      cp16_g2l(Qh + base + (size_t)(sBase + r)*DHH + co, &Qs[r][co]);
    }
  }

  float m_run[8], l_run[8];
#pragma unroll
  for (int i = 0; i < 8; ++i) { m_run[i] = -1e30f; l_run[i] = 0.0f; }
  v8f oacc[4] = {};

  for (int kb = 0; kb < SS/64; ++kb) {
    const int kBase = kb * 64;
    __syncthreads();                 // previous iteration done reading Ks/Vt
    { // stage K tile (64x64) and V^T tile (64x64): 512 chunks each, 2/thread
#pragma unroll
      for (int j = 0; j < 2; ++j) {
        const int cc = tid + 256*j;
        const int r = cc >> 3, co = (cc & 7) * 8;
        cp16_g2l(Kh  + base  + (size_t)(kBase + r)*DHH + co, &Ks[r][co]);
        cp16_g2l(VTh + vbase + (size_t)r*SS + kBase + co,    &Vt[r][co]);
      }
    }
    async_wait();
    __syncthreads();

    // S tile = Q K^T : 16x64 per wave, K-dim = DH = 64 -> 2 WMMA steps.
    v8f sacc[4] = {};
#pragma unroll
    for (int ks = 0; ks < 64; ks += 32) {
      v8h alo = *(const v8h*)&Qs[w*16 + ml][ks + half*8];
      v8h ahi = *(const v8h*)&Qs[w*16 + ml][ks + 16 + half*8];
      v16h a  = join8(alo, ahi);
#pragma unroll
      for (int nT = 0; nT < 4; ++nT) {
        v8h blo = *(const v8h*)&Ks[nT*16 + ml][ks + half*8];
        v8h bhi = *(const v8h*)&Ks[nT*16 + ml][ks + 16 + half*8];
        sacc[nT] = wmma16(a, join8(blo, bhi), sacc[nT]);
      }
    }

    // Scale, LSA diagonal mask, online softmax (row = one 16-lane group).
#pragma unroll
    for (int i = 0; i < 8; ++i) {
      const int qRow = sBase + w*16 + half*8 + i;
      float mx = m_run[i];
#pragma unroll
      for (int nT = 0; nT < 4; ++nT) {
        float s = sacc[nT][i] * 0.125f;                 // 1/sqrt(DH)
        if (qRow == kBase + nT*16 + ml) s = LSA_NEG;    // mask diagonal
        sacc[nT][i] = s;
        mx = fmaxf(mx, s);
      }
#pragma unroll
      for (int off = 1; off < 16; off <<= 1)
        mx = fmaxf(mx, __shfl_xor(mx, off, 32));
      const float alpha = __expf(m_run[i] - mx);
      m_run[i] = mx;
      float rs = 0.0f;
#pragma unroll
      for (int nT = 0; nT < 4; ++nT) {
        const float p = __expf(sacc[nT][i] - mx);
        rs += p;
        Ps[w*16 + half*8 + i][nT*16 + ml] = (_Float16)p; // C-layout -> LDS
      }
#pragma unroll
      for (int off = 1; off < 16; off <<= 1)
        rs += __shfl_xor(rs, off, 32);
      l_run[i] = l_run[i] * alpha + rs;
#pragma unroll
      for (int dT = 0; dT < 4; ++dT) oacc[dT][i] *= alpha;
    }

    // O += P V. Ps rows are wave-private; per-wave LDS ops are in-order.
#pragma unroll
    for (int ks = 0; ks < 64; ks += 32) {
      v8h alo = *(const v8h*)&Ps[w*16 + ml][ks + half*8];
      v8h ahi = *(const v8h*)&Ps[w*16 + ml][ks + 16 + half*8];
      v16h a  = join8(alo, ahi);
#pragma unroll
      for (int dT = 0; dT < 4; ++dT) {
        v8h blo = *(const v8h*)&Vt[dT*16 + ml][ks + half*8];
        v8h bhi = *(const v8h*)&Vt[dT*16 + ml][ks + 16 + half*8];
        oacc[dT] = wmma16(a, join8(blo, bhi), oacc[dT]);
      }
    }
  }

  // Normalize and store merged-head layout [B,S,D] as f16 for the out-proj.
#pragma unroll
  for (int dT = 0; dT < 4; ++dT) {
    const int col = h*DHH + dT*16 + ml;
#pragma unroll
    for (int i = 0; i < 8; ++i) {
      const int srow = sBase + w*16 + half*8 + i;
      Aout[((size_t)b*SS + srow)*DD + col] = (_Float16)(oacc[dT][i] / l_run[i]);
    }
  }
}

// ---------------------------------------------------------------------------
// Kernel 3: out = A @ Wo^T + bo, A f16 [M, D] from workspace, fp32 output.
// Same 128x128 tiling as kernel 1.
// ---------------------------------------------------------------------------
__global__ __launch_bounds__(256)
void out_proj_kernel(const _Float16* __restrict__ Xh, const float* __restrict__ W,
                     const float* __restrict__ bias, float* __restrict__ Out)
{
  __shared__ _Float16 As[128][40];
  __shared__ _Float16 Bs[128][40];

  const int tid  = threadIdx.x;
  const int lane = tid & 31;
  const int w    = tid >> 5;
  const int wr   = w >> 1;
  const int wc   = w & 1;
  const int half = lane >> 4;
  const int ml   = lane & 15;
  const int mBase = blockIdx.x * 128;
  const int nBase = blockIdx.y * 128;

  v8f acc[2][4] = {};
  const int rS = tid >> 1;
  const int kC = (tid & 1) * 16;

  for (int k0 = 0; k0 < DD; k0 += 32) {
    const _Float16* xs = Xh + (size_t)(mBase + rS) * DD + k0 + kC;
    const float*    ws = W  + (size_t)(nBase + rS) * DD + k0 + kC;
#pragma unroll
    for (int j = 0; j < 16; j += 8)
      *(v8h*)&As[rS][kC + j] = *(const v8h*)&xs[j];
#pragma unroll
    for (int j = 0; j < 16; ++j) Bs[rS][kC + j] = (_Float16)ws[j];
    if (k0 + 32 < DD) {
      __builtin_prefetch(xs + 32, 0, 1);
      __builtin_prefetch(ws + 32, 0, 1);
    }
    __syncthreads();

    v16h a[2];
#pragma unroll
    for (int mT = 0; mT < 2; ++mT) {
      v8h alo = *(const v8h*)&As[wr*32 + mT*16 + ml][half*8];
      v8h ahi = *(const v8h*)&As[wr*32 + mT*16 + ml][16 + half*8];
      a[mT] = join8(alo, ahi);
    }
#pragma unroll
    for (int nT = 0; nT < 4; ++nT) {
      v8h blo = *(const v8h*)&Bs[wc*64 + nT*16 + ml][half*8];
      v8h bhi = *(const v8h*)&Bs[wc*64 + nT*16 + ml][16 + half*8];
      v16h bf = join8(blo, bhi);
#pragma unroll
      for (int mT = 0; mT < 2; ++mT)
        acc[mT][nT] = wmma16(a[mT], bf, acc[mT][nT]);
    }
    __syncthreads();
  }

#pragma unroll
  for (int nT = 0; nT < 4; ++nT) {
    const int col = nBase + wc*64 + nT*16 + ml;
    const float bv = bias[col];
#pragma unroll
    for (int mT = 0; mT < 2; ++mT) {
#pragma unroll
      for (int i = 0; i < 8; ++i) {
        const int row = mBase + wr*32 + mT*16 + half*8 + i;
        Out[(size_t)row*DD + col] = acc[mT][nT][i] + bv;
      }
    }
  }
}

// ---------------------------------------------------------------------------
extern "C" void kernel_launch(void* const* d_in, const int* in_sizes, int n_in,
                              void* d_out, int out_size, void* d_ws, size_t ws_size,
                              hipStream_t stream) {
  (void)in_sizes; (void)n_in; (void)out_size; (void)ws_size;
  const float* x_q  = (const float*)d_in[0];
  const float* x_kv = (const float*)d_in[1];
  const float* Wq   = (const float*)d_in[2];
  const float* bq   = (const float*)d_in[3];
  const float* Wk   = (const float*)d_in[4];
  const float* bk   = (const float*)d_in[5];
  const float* Wv   = (const float*)d_in[6];
  const float* bv   = (const float*)d_in[7];
  const float* Wo   = (const float*)d_in[8];
  const float* bo   = (const float*)d_in[9];
  float* out = (float*)d_out;

  const size_t elems = (size_t)BB * SS * DD;   // 8,388,608
  _Float16* Qh  = (_Float16*)d_ws;             // needs 4*elems*2 = 64 MB of ws
  _Float16* Kh  = Qh + elems;
  _Float16* VTh = Kh + elems;                  // transposed [B,H,DH,S]
  _Float16* Ah  = VTh + elems;

  dim3 gGemm(BB*SS/128, DD/128);               // 64 x 8
  qkv_proj_kernel<<<gGemm, 256, 0, stream>>>(x_q,  Wq, bq, Qh,  0);
  qkv_proj_kernel<<<gGemm, 256, 0, stream>>>(x_kv, Wk, bk, Kh,  0);
  qkv_proj_kernel<<<gGemm, 256, 0, stream>>>(x_kv, Wv, bv, VTh, 1);
  lsa_attn_kernel<<<BB*HH*(SS/128), 256, 0, stream>>>(Qh, Kh, VTh, Ah);
  out_proj_kernel<<<gGemm, 256, 0, stream>>>(Ah, Wo, bo, out);
}
